// SingleHeadAttention_18794776887966
// MI455X (gfx1250) — matrix operand order repaired and checked by reading, not compiled, not run on hardware
//
#include <hip/hip_runtime.h>

typedef __attribute__((ext_vector_type(16))) _Float16 v16h;
typedef __attribute__((ext_vector_type(8)))  _Float16 v8h;
typedef __attribute__((ext_vector_type(8)))  float    v8f;
typedef __attribute__((ext_vector_type(4)))  float    v4f;

#define BATCH 4
#define SEQ   4096
#define EMB   512
#define HD    64
#define ROWS  (BATCH * SEQ)          // 16384
#define WELT  (3 * EMB * HD)         // 98304 weight elements

__device__ __forceinline__ v8f wmma_f16(v16h a, v16h b, v8f c) {
  return __builtin_amdgcn_wmma_f32_16x16x32_f16(
      false, a, false, b, (short)0, c, false, false);
}

__device__ __forceinline__ v16h cat8(v8h a, v8h b) {
  return __builtin_shufflevector(a, b, 0,1,2,3,4,5,6,7,8,9,10,11,12,13,14,15);
}

// ---- fragment loaders ------------------------------------------------------
// A fragment (16x32, ISA layout) from an f16 plane, row-major [rows][ld]:
// lane L: row m=L&15; half j -> K = kb + ((j&8)<<1) + (L<16?0:8) + (j&7)
// => two contiguous 8-half (16B) runs -> two b128 loads.
__device__ __forceinline__ v16h load_a_f16(const _Float16* base, int ld, int kb) {
  const int lane = threadIdx.x & 31;
  const int m    = lane & 15;
  const int kh   = (lane < 16) ? 0 : 8;
  const _Float16* row = base + (size_t)m * ld + kb + kh;
  return cat8(*(const v8h*)(row), *(const v8h*)(row + 16));
}

// B fragment (32x16, ISA layout) from an f16 plane stored so K is contiguous:
// element (K,N) at base[N*ld + K]; lane L: N=L&15, K = kb + ((j&8)<<1) + (L<16?0:8) + (j&7)
// => two contiguous 8-half (16B) runs -> two b128 loads.
__device__ __forceinline__ v16h load_b_f16(const _Float16* base, size_t ld, int kb) {
  const int lane = threadIdx.x & 31;
  const int n    = lane & 15;
  const int k0   = (lane < 16) ? 0 : 8;
  const _Float16* p = base + (size_t)n * ld + kb + k0;
  return cat8(*(const v8h*)(p), *(const v8h*)(p + 16));
}

// A fragment from fp32 memory with on-the-fly hi/lo split (projection only).
__device__ __forceinline__ void load_a_f32cvt(const float* base, int ld, int kb,
                                              v16h& ah, v16h& al) {
  const int lane = threadIdx.x & 31;
  const int m    = lane & 15;
  const int kh   = (lane < 16) ? 0 : 8;
  const float* row = base + (size_t)m * ld + kb + kh;
  float x[16];
  *(v4f*)(x + 0)  = *(const v4f*)(row + 0);
  *(v4f*)(x + 4)  = *(const v4f*)(row + 4);
  *(v4f*)(x + 8)  = *(const v4f*)(row + 16);
  *(v4f*)(x + 12) = *(const v4f*)(row + 20);
#pragma unroll
  for (int j = 0; j < 16; ++j) {
    _Float16 h = (_Float16)x[j];
    ah[j] = h;
    al[j] = (_Float16)(x[j] - (float)h);
  }
}

__device__ __forceinline__ float rmax16(float x) {
#pragma unroll
  for (int m = 8; m >= 1; m >>= 1) x = fmaxf(x, __shfl_xor(x, m, 32));
  return x;
}
__device__ __forceinline__ float rsum16(float x) {
#pragma unroll
  for (int m = 8; m >= 1; m >>= 1) x += __shfl_xor(x, m, 32);
  return x;
}

// ---------------------------------------------------------------------------
// Kernel 0: weight prep — transpose + hi/lo split Wq|Wk|Wv into f16 planes
// Wth/Wtl [3][HD][EMB] so projection B-fragments are contiguous b128 loads.
// ---------------------------------------------------------------------------
__global__ __launch_bounds__(256) void wprep_kernel(
    const float* __restrict__ Wq, const float* __restrict__ Wk,
    const float* __restrict__ Wv, _Float16* __restrict__ Wth,
    _Float16* __restrict__ Wtl) {
  const int idx = blockIdx.x * 256 + threadIdx.x;     // [0, 98304)
  const int mat = idx >> 15;                          // EMB*HD = 32768 per matrix
  const int rem = idx & 32767;
  const int k   = rem >> 6;                           // [0, 512)
  const int n   = rem & 63;                           // [0, 64)
  const float* W = (mat == 0) ? Wq : (mat == 1) ? Wk : Wv;
  const float x = W[(size_t)k * HD + n];
  const _Float16 h = (_Float16)x;
  const size_t o = (size_t)mat * (EMB * HD) + (size_t)n * EMB + k;
  Wth[o] = h;
  Wtl[o] = (_Float16)(x - (float)h);
}

// ---------------------------------------------------------------------------
// Kernel 1: fused QKV projection.  One wave per 16 rows, all 12 col tiles in
// registers (embedded read once from HBM; weights L2-resident).  Emits
// pre-split f16 planes: Qh/Ql,Kh/Kl row-major [ROWS][HD]; Vth/Vtl transposed
// [HD][ROWS].  Q pre-scaled by 1/sqrt(HD)=0.125.
// ---------------------------------------------------------------------------
__global__ __launch_bounds__(256) void qkv_kernel(
    const float* __restrict__ emb,
    const _Float16* __restrict__ Wth, const _Float16* __restrict__ Wtl,
    _Float16* __restrict__ Qh, _Float16* __restrict__ Ql,
    _Float16* __restrict__ Kh, _Float16* __restrict__ Kl,
    _Float16* __restrict__ Vth, _Float16* __restrict__ Vtl) {
  const int lane  = threadIdx.x & 31;
  const int w     = threadIdx.x >> 5;
  const int n     = lane & 15;
  const int hi8   = (lane < 16) ? 0 : 8;
  const int mBase = (blockIdx.x * 8 + w) * 16;

  v8f acc[12];
#pragma unroll
  for (int i = 0; i < 12; ++i) acc[i] = (v8f){0.f,0.f,0.f,0.f,0.f,0.f,0.f,0.f};

  for (int kb = 0; kb < EMB; kb += 32) {
    v16h ah, al;
    load_a_f32cvt(emb + (size_t)mBase * EMB, EMB, kb, ah, al);
#pragma unroll
    for (int mt = 0; mt < 3; ++mt) {
#pragma unroll
      for (int t = 0; t < 4; ++t) {
        const size_t wo = (size_t)mt * (EMB * HD) + (size_t)(t * 16) * EMB;
        const v16h bh = load_b_f16(Wth + wo, EMB, kb);
        const v16h bl = load_b_f16(Wtl + wo, EMB, kb);
        const int i = mt * 4 + t;
        acc[i] = wmma_f16(ah, bh, acc[i]);
        acc[i] = wmma_f16(ah, bl, acc[i]);
        acc[i] = wmma_f16(al, bh, acc[i]);
      }
    }
  }

#pragma unroll
  for (int mt = 0; mt < 3; ++mt) {
    const float sc = (mt == 0) ? 0.125f : 1.0f;
#pragma unroll
    for (int t = 0; t < 4; ++t)
#pragma unroll
      for (int r = 0; r < 8; ++r) {
        const int row = mBase + r + hi8;
        const int col = t * 16 + n;
        const float v = acc[mt * 4 + t][r] * sc;
        const _Float16 h = (_Float16)v;
        const _Float16 l = (_Float16)(v - (float)h);
        if (mt == 0)      { Qh[(size_t)row * HD + col] = h;  Ql[(size_t)row * HD + col] = l; }
        else if (mt == 1) { Kh[(size_t)row * HD + col] = h;  Kl[(size_t)row * HD + col] = l; }
        else              { Vth[(size_t)col * ROWS + row] = h; Vtl[(size_t)col * ROWS + row] = l; }
      }
  }
}

// ---------------------------------------------------------------------------
// Kernel 2: causal flash attention.  One wave = 16 query rows, 32-key chunks
// up to the diagonal (wave-uniform bound -> EXEC all-1s for WMMA).  All K/V/Q
// operand loads are pre-split f16 b128 loads; only P is split here, staged
// through per-wave f16 LDS planes ordered with s_wait_dscnt.
// ---------------------------------------------------------------------------
__global__ __launch_bounds__(256) void attn_kernel(
    const _Float16* __restrict__ Qh, const _Float16* __restrict__ Ql,
    const _Float16* __restrict__ Kh, const _Float16* __restrict__ Kl,
    const _Float16* __restrict__ Vth, const _Float16* __restrict__ Vtl,
    float* __restrict__ out) {
  __shared__ _Float16 pldh[8][16 * 40];   // 16 rows x 32 cols, stride 40 (80B, 16B-aligned)
  __shared__ _Float16 pldl[8][16 * 40];

  const int lane  = threadIdx.x & 31;
  const int w     = threadIdx.x >> 5;
  const int n     = lane & 15;
  const int hi8   = (lane < 16) ? 0 : 8;
  const int b     = blockIdx.y;
  const int qBase = blockIdx.x * 128 + w * 16;
  const int bT    = b * SEQ;

  v16h qh[2], ql[2];
#pragma unroll
  for (int kf = 0; kf < 2; ++kf) {
    qh[kf] = load_a_f16(Qh + (size_t)(bT + qBase) * HD, HD, kf * 32);
    ql[kf] = load_a_f16(Ql + (size_t)(bT + qBase) * HD, HD, kf * 32);
  }

  v8f acc[4];
#pragma unroll
  for (int t = 0; t < 4; ++t) acc[t] = (v8f){0.f,0.f,0.f,0.f,0.f,0.f,0.f,0.f};
  float m_r[8], l_r[8];
#pragma unroll
  for (int r = 0; r < 8; ++r) { m_r[r] = -1e30f; l_r[r] = 0.f; }

  for (int jb = 0; jb < qBase + 16; jb += 32) {
    // ---- S = Q K^T : two 16x16 tiles, 3-way split each ----
    v8f S[2];
#pragma unroll
    for (int t = 0; t < 2; ++t) {
      S[t] = (v8f){0.f,0.f,0.f,0.f,0.f,0.f,0.f,0.f};
      const size_t ko = (size_t)(bT + jb + t * 16) * HD;
#pragma unroll
      for (int kf = 0; kf < 2; ++kf) {
        const v16h kbh = load_b_f16(Kh + ko, HD, kf * 32);
        const v16h kbl = load_b_f16(Kl + ko, HD, kf * 32);
        S[t] = wmma_f16(qh[kf], kbh, S[t]);
        S[t] = wmma_f16(qh[kf], kbl, S[t]);
        S[t] = wmma_f16(ql[kf], kbh, S[t]);
      }
    }

    // ---- causal mask + online softmax (fp32) ----
    float csc[8];
#pragma unroll
    for (int r = 0; r < 8; ++r) {
      const int q = qBase + r + hi8;
#pragma unroll
      for (int t = 0; t < 2; ++t)
        if (jb + t * 16 + n > q) S[t][r] = -1e30f;
      const float mnew = fmaxf(m_r[r], rmax16(fmaxf(S[0][r], S[1][r])));
      csc[r] = __expf(m_r[r] - mnew);
      const float p0 = __expf(S[0][r] - mnew);
      const float p1 = __expf(S[1][r] - mnew);
      S[0][r] = p0; S[1][r] = p1;
      l_r[r] = l_r[r] * csc[r] + rsum16(p0 + p1);
      m_r[r] = mnew;
    }
#pragma unroll
    for (int t = 0; t < 4; ++t)
#pragma unroll
      for (int r = 0; r < 8; ++r) acc[t][r] *= csc[r];

    // ---- split P in registers, C-layout -> A-layout via f16 LDS planes ----
#pragma unroll
    for (int t = 0; t < 2; ++t)
#pragma unroll
      for (int r = 0; r < 8; ++r) {
        const float p = S[t][r];
        const _Float16 h = (_Float16)p;
        const int idx = (r + hi8) * 40 + t * 16 + n;
        pldh[w][idx] = h;
        pldl[w][idx] = (_Float16)(p - (float)h);
      }
    asm volatile("s_wait_dscnt 0x0" ::: "memory");   // same-wave store->load order
    const v16h ph = load_a_f16(&pldh[w][0], 40, 0);
    const v16h pl = load_a_f16(&pldl[w][0], 40, 0);

    // ---- O += P V : K=32 keys, 4 N-tiles, 3-way split ----
#pragma unroll
    for (int t = 0; t < 4; ++t) {
      const size_t vo = (size_t)(t * 16) * ROWS + bT + jb;
      const v16h vbh = load_b_f16(Vth + vo, ROWS, 0);
      const v16h vbl = load_b_f16(Vtl + vo, ROWS, 0);
      acc[t] = wmma_f16(ph, vbh, acc[t]);
      acc[t] = wmma_f16(ph, vbl, acc[t]);
      acc[t] = wmma_f16(pl, vbh, acc[t]);
    }
  }

  // ---- finalize: O / l ----
#pragma unroll
  for (int r = 0; r < 8; ++r) {
    const float inv = 1.0f / l_r[r];
    const size_t row = (size_t)(bT + qBase + r + hi8);
#pragma unroll
    for (int t = 0; t < 4; ++t)
      out[row * HD + t * 16 + n] = acc[t][r] * inv;
  }
}

extern "C" void kernel_launch(void* const* d_in, const int* in_sizes, int n_in,
                              void* d_out, int out_size, void* d_ws, size_t ws_size,
                              hipStream_t stream) {
  (void)in_sizes; (void)n_in; (void)out_size; (void)ws_size;
  const float* emb = (const float*)d_in[0];
  const float* Wq  = (const float*)d_in[1];
  const float* Wk  = (const float*)d_in[2];
  const float* Wv  = (const float*)d_in[3];

  const size_t R = (size_t)ROWS * HD;            // 1,048,576 halves per plane
  _Float16* p   = (_Float16*)d_ws;               // ~13 MB total
  _Float16* Qhp = p;            _Float16* Qlp = p + R;
  _Float16* Khp = p + 2 * R;    _Float16* Klp = p + 3 * R;
  _Float16* Vth = p + 4 * R;    _Float16* Vtl = p + 5 * R;
  _Float16* Wth = p + 6 * R;    _Float16* Wtl = Wth + WELT;

  wprep_kernel<<<WELT / 256, 256, 0, stream>>>(Wq, Wk, Wv, Wth, Wtl);
  qkv_kernel<<<128, 256, 0, stream>>>(emb, Wth, Wtl, Qhp, Qlp, Khp, Klp, Vth, Vtl);
  attn_kernel<<<dim3(32, 4), 256, 0, stream>>>(Qhp, Qlp, Khp, Klp, Vth, Vtl,
                                               (float*)d_out);
}
